// MultiHeadAttention_36730560315858
// MI455X (gfx1250) — matrix operand order repaired and checked by reading, not compiled, not run on hardware
//
#include <hip/hip_runtime.h>
#include <hip/hip_bf16.h>
#include <math.h>

typedef __attribute__((ext_vector_type(16))) __bf16 v16bf;
typedef __attribute__((ext_vector_type(8)))  __bf16 v8bf;
typedef __attribute__((ext_vector_type(8)))  float  v8f;

namespace {
constexpr int kB    = 64;
constexpr int kSeq  = 210;   // 15*14 spatial sites
constexpr int kSeqP = 224;   // padded to 14*16
constexpr int kTok  = kB * kSeq;   // 13440 (divisible by 16)
constexpr int kEmb  = 512;
constexpr int kNH   = 8;
constexpr int kHD   = 64;
constexpr int kMT   = kTok / 16;   // 840 row tiles
constexpr int kNT   = kEmb / 16;   // 32 col tiles
constexpr float kScale = 0.125f;   // (4*4*4)^-0.5
}

__device__ __forceinline__ v16bf pack16(v8bf lo, v8bf hi) {
  v16bf r;
#pragma unroll
  for (int j = 0; j < 8; ++j) { r[j] = lo[j]; r[j + 8] = hi[j]; }
  return r;
}

__device__ __forceinline__ v8f bwmma(v16bf a, v16bf b, v8f c) {
  // D = A(16x32 bf16) * B(32x16 bf16) + C(16x16 f32)
  return __builtin_amdgcn_wmma_f32_16x16x32_bf16(false, a, false, b, (short)0, c,
                                                 false, false);
}

// ---------------------------------------------------------------- stage 1a
__global__ void k_cvt_x(const float* __restrict__ x, __bf16* __restrict__ xb, int n) {
  int i = blockIdx.x * blockDim.x + threadIdx.x;
  if (i < n) xb[i] = (__bf16)x[i];
}

// ---------------------------------------------------------------- stage 1b
// Build Kronecker weights W0 (x) W1 (x) W2 as 512x512, stored directly in
// WMMA B-fragment order: element index = ((kt*32 + nt)*512 + lane*16 + j),
// with n = nt*16 + (lane&15), k = kt*32 + (lane>>4)*8 + (j&7) + (j>>3)*16.
struct WArgs {
  const float* w0[4];
  const float* w1[4];
  const float* w2[4];
};

__global__ void k_build_wt(WArgs wa, __bf16* __restrict__ wt) {
  int i = blockIdx.x * blockDim.x + threadIdx.x;   // [0, 4*512*512)
  int p    = i >> 18;
  int rem  = i & 0x3FFFF;
  int frag = rem >> 9;           // kt*32 + nt
  int lj   = rem & 511;          // lane*16 + j
  int lane = lj >> 4, j = lj & 15;
  int kt = frag >> 5, nt = frag & 31;
  int n  = nt * 16 + (lane & 15);
  int kk = kt * 32 + (lane >> 4) * 8 + (j & 7) + ((j >> 3) << 4);
  int a  = n >> 6, c = (n >> 3) & 7, d = n & 7;       // output (e0,e1,e2)
  int xx = kk >> 6, yy = (kk >> 3) & 7, zz = kk & 7;  // input  (e0,e1,e2)
  float v = wa.w0[p][a * 8 + xx] * wa.w1[p][c * 8 + yy] * wa.w2[p][d * 8 + zz];
  wt[i] = (__bf16)v;
}

// ---------------------------------------------------------------- GEMM mainloop
// A: row-major bf16 [.,512]; per-lane row = tile_row + (lane&15).
// A-frag element j: k = kt*32 + (j&7) + ((j>>3)<<4) + (hi<<3)  -> two 16B loads.
// B: pre-swizzled fragments -> one contiguous 32B load per lane.
__device__ __forceinline__ v8f gemm512(const __bf16* __restrict__ arow,
                                       const __bf16* __restrict__ wlane, int hi) {
  v8f acc = {};
#pragma unroll
  for (int kt = 0; kt < 16; ++kt) {
    if (kt < 15) __builtin_prefetch((const void*)(wlane + (kt + 1) * (32 * 512)), 0, 0);
    v8bf a0 = *(const v8bf*)(arow + kt * 32 + hi * 8);
    v8bf a1 = *(const v8bf*)(arow + kt * 32 + 16 + hi * 8);
    v16bf b = *(const v16bf*)(wlane + kt * (32 * 512));
    acc = bwmma(pack16(a0, a1), b, acc);
  }
  return acc;
}

// ---------------------------------------------------------------- stage 2: QKV
__global__ __launch_bounds__(32) void k_proj_qkv(
    const __bf16* __restrict__ xb, const __bf16* __restrict__ wt,
    const float* __restrict__ bq, const float* __restrict__ bk,
    const float* __restrict__ bv,
    __bf16* __restrict__ Qb, __bf16* __restrict__ Kb, __bf16* __restrict__ Vt) {
  int mt = blockIdx.x, nt = blockIdx.y, pj = blockIdx.z;
  int lane = threadIdx.x, r = lane & 15, hi = lane >> 4;
  const __bf16* arow  = xb + (size_t)(mt * 16 + r) * kEmb;
  const __bf16* wlane = wt + (size_t)pj * (kEmb * kEmb) + (size_t)nt * 512 + lane * 16;
  v8f acc = gemm512(arow, wlane, hi);

  int n = nt * 16 + r;
  const float* bias = (pj == 0) ? bq : (pj == 1) ? bk : bv;
  float bz = bias[n];
  // channel -> (head triple, head-dim):  e0 = xi*2+h1 etc.
  int e0 = n >> 6, e1 = (n >> 3) & 7, e2 = n & 7;
  int h = ((e0 & 1) << 2) | ((e1 & 1) << 1) | (e2 & 1);
  int d = ((e0 >> 1) << 4) | ((e1 >> 1) << 2) | (e2 >> 1);
#pragma unroll
  for (int g = 0; g < 8; ++g) {
    int m = g + hi * 8;
    int t = mt * 16 + m;
    int bb = t / kSeq, s = t - bb * kSeq;
    float val = acc[g] + bz;
    size_t bh = (size_t)bb * kNH + h;
    if (pj == 0)      Qb[(bh * kSeqP + s) * kHD + d] = (__bf16)val;
    else if (pj == 1) Kb[(bh * kSeqP + s) * kHD + d] = (__bf16)val;
    else              Vt[(bh * kHD + d) * kSeqP + s] = (__bf16)val;   // transposed
  }
}

// ---------------------------------------------------------------- stage 3: attention
__global__ __launch_bounds__(32) void k_attn(
    const __bf16* __restrict__ Qb, const __bf16* __restrict__ Kb,
    const __bf16* __restrict__ Vt, __bf16* __restrict__ Ob) {
  __shared__ float  sc[16 * kSeqP];   // logits / exp
  __shared__ __bf16 pr[16 * kSeqP];   // probabilities
  int bh = blockIdx.x;                // batch*8 + head
  int qt = blockIdx.y;                // query tile 0..13
  int lane = threadIdx.x, r = lane & 15, hi = lane >> 4;
  const __bf16* Q = Qb + (size_t)bh * kSeqP * kHD;
  const __bf16* K = Kb + (size_t)bh * kSeqP * kHD;
  const __bf16* V = Vt + (size_t)bh * kHD * kSeqP;

  // Q fragments: 16x64 = two 16x32 A tiles
  const __bf16* qrow = Q + (size_t)(qt * 16 + r) * kHD;
  v16bf aq0 = pack16(*(const v8bf*)(qrow + hi * 16),      *(const v8bf*)(qrow + 8 + hi * 16));
  v16bf aq1 = pack16(*(const v8bf*)(qrow + 32 + hi * 16), *(const v8bf*)(qrow + 40 + hi * 16));

  // scores = Q K^T * scale  (B-frag of K^T: lane -> key row, contiguous in k)
  for (int jt = 0; jt < 14; ++jt) {
    const __bf16* krow = K + (size_t)(jt * 16 + r) * kHD;
    v16bf bk0 = *(const v16bf*)(krow + hi * 16);
    v16bf bk1 = *(const v16bf*)(krow + 32 + hi * 16);
    v8f acc = {};
    acc = bwmma(aq0, bk0, acc);
    acc = bwmma(aq1, bk1, acc);
    int col = jt * 16 + r;
    float mask = (col < kSeq) ? 0.f : -1e30f;
#pragma unroll
    for (int g = 0; g < 8; ++g) {
      int m = g + hi * 8;
      sc[m * kSeqP + col] = acc[g] * kScale + mask;
    }
  }
  __syncthreads();

  // softmax: two lanes per row (column halves), combine via shfl_xor(16)
  int c0 = hi * 112;
  float mx = -1e30f;
  for (int c = 0; c < 112; ++c) mx = fmaxf(mx, sc[r * kSeqP + c0 + c]);
  mx = fmaxf(mx, __shfl_xor(mx, 16, 32));
  float sum = 0.f;
  for (int c = 0; c < 112; ++c) {
    float e = __expf(sc[r * kSeqP + c0 + c] - mx);
    sc[r * kSeqP + c0 + c] = e;
    sum += e;
  }
  sum += __shfl_xor(sum, 16, 32);
  float inv = 1.0f / sum;
  for (int c = 0; c < 112; ++c)
    pr[r * kSeqP + c0 + c] = (__bf16)(sc[r * kSeqP + c0 + c] * inv);
  __syncthreads();

  // O = A * V  (A from LDS probs; V^T gives contiguous B-frag loads)
  int b = bh >> 3, h = bh & 7;
#pragma unroll
  for (int ntk = 0; ntk < 4; ++ntk) {
    v8f acc = {};
#pragma unroll
    for (int kt = 0; kt < 7; ++kt) {
      const __bf16* pp = pr + r * kSeqP + kt * 32;
      v16bf a = pack16(*(const v8bf*)(pp + hi * 16), *(const v8bf*)(pp + 8 + hi * 16));
      const __bf16* vp = V + (size_t)(ntk * 16 + r) * kSeqP + kt * 32 + hi * 16;
      v16bf bvv = *(const v16bf*)vp;
      acc = bwmma(a, bvv, acc);
    }
    int d = ntk * 16 + r;
    int e0 = ((d >> 4) & 3) * 2 + (h >> 2);
    int e1 = ((d >> 2) & 3) * 2 + ((h >> 1) & 1);
    int e2 = (d & 3) * 2 + (h & 1);
    int ch = (e0 << 6) | (e1 << 3) | e2;
#pragma unroll
    for (int g = 0; g < 8; ++g) {
      int s = qt * 16 + g + hi * 8;
      if (s < kSeq) {
        size_t t = (size_t)b * kSeq + s;
        Ob[t * kEmb + ch] = (__bf16)acc[g];
      }
    }
  }
}

// ---------------------------------------------------------------- stage 4: output proj
__global__ __launch_bounds__(32) void k_proj_out(
    const __bf16* __restrict__ ob, const __bf16* __restrict__ wt,
    const float* __restrict__ bo, float* __restrict__ out) {
  int mt = blockIdx.x, nt = blockIdx.y;
  int lane = threadIdx.x, r = lane & 15, hi = lane >> 4;
  const __bf16* arow  = ob + (size_t)(mt * 16 + r) * kEmb;
  const __bf16* wlane = wt + (size_t)3 * (kEmb * kEmb) + (size_t)nt * 512 + lane * 16;
  v8f acc = gemm512(arow, wlane, hi);
  int n = nt * 16 + r;
  float bz = bo[n];
#pragma unroll
  for (int g = 0; g < 8; ++g) {
    int t = mt * 16 + g + hi * 8;
    out[(size_t)t * kEmb + n] = acc[g] + bz;
  }
}

// ---------------------------------------------------------------- launcher
extern "C" void kernel_launch(void* const* d_in, const int* in_sizes, int n_in,
                              void* d_out, int out_size, void* d_ws, size_t ws_size,
                              hipStream_t stream) {
  (void)in_sizes; (void)n_in; (void)out_size; (void)ws_size;
  const float* x = (const float*)d_in[0];

  char* ws = (char*)d_ws;
  size_t off = 0;
  __bf16* Xb = (__bf16*)(ws + off); off += (size_t)kTok * kEmb * 2;            // 13.8 MB
  __bf16* WT = (__bf16*)(ws + off); off += (size_t)4 * kEmb * kEmb * 2;        //  2.1 MB
  __bf16* Qb = (__bf16*)(ws + off); off += (size_t)kB * kNH * kSeqP * kHD * 2; // 14.7 MB
  __bf16* Kb = (__bf16*)(ws + off); off += (size_t)kB * kNH * kSeqP * kHD * 2;
  __bf16* Vt = (__bf16*)(ws + off); off += (size_t)kB * kNH * kSeqP * kHD * 2;
  __bf16* Ob = (__bf16*)(ws + off); off += (size_t)kTok * kEmb * 2;

  int nx = kTok * kEmb;
  k_cvt_x<<<dim3((nx + 255) / 256), dim3(256), 0, stream>>>(x, Xb, nx);

  WArgs wa;
  for (int p = 0; p < 4; ++p) {
    wa.w0[p] = (const float*)d_in[1 + p * 4 + 0];
    wa.w1[p] = (const float*)d_in[1 + p * 4 + 1];
    wa.w2[p] = (const float*)d_in[1 + p * 4 + 2];
  }
  k_build_wt<<<dim3((4 * kEmb * kEmb) / 256), dim3(256), 0, stream>>>(wa, WT);

  k_proj_qkv<<<dim3(kMT, kNT, 3), dim3(32), 0, stream>>>(
      Xb, WT, (const float*)d_in[4], (const float*)d_in[8], (const float*)d_in[12],
      Qb, Kb, Vt);

  k_attn<<<dim3(kB * kNH, 14), dim3(32), 0, stream>>>(Qb, Kb, Vt, Ob);

  k_proj_out<<<dim3(kMT, kNT), dim3(32), 0, stream>>>(
      Ob, WT, (const float*)d_in[16], (float*)d_out);
}
